// ArcFaceLoss_62706522521602
// MI455X (gfx1250) — matrix operand order repaired and checked by reading, not compiled, not run on hardware
//
#include <hip/hip_runtime.h>

#define BATCH 512
#define EMB   512
#define NCLS  100000

// split-K staging for the in-kernel normalized weight tile
#define KC   256
#define LDK  (KC + 8)   // +8 bf16 (=16B) row pad -> conflict-free ds_load_b128

typedef __attribute__((ext_vector_type(16))) __bf16 v16bf;
typedef __attribute__((ext_vector_type(8)))  __bf16 v8bf;
typedef __attribute__((ext_vector_type(8)))  float  v8f;

// ArcFace constants (m = 0.5, s = 30)
#define ARC_S   30.0f
#define ARC_CM  0.87758256189037271612f     // cos(m)
#define ARC_SM  0.47942553860420300027f     // sin(m)
#define ARC_TH  (-0.87758256189037271612f)  // cos(pi - m)
#define ARC_MM  0.23971276930210150013f     // sin(pi - m) * m

__device__ __forceinline__ float wave_reduce_sum(float v) {
#pragma unroll
    for (int off = 16; off > 0; off >>= 1) v += __shfl_xor(v, off, 32);
    return v;
}

// ---------------------------------------------------------------------------
// Kernel 1: L2-normalize embeddings rows, split into bf16 hi/lo planes.
// One wave per row; 64 blocks x 256 threads cover 512 rows.
// ---------------------------------------------------------------------------
__global__ __launch_bounds__(256)
void norm_split_emb(const float* __restrict__ emb,
                    __bf16* __restrict__ ahi, __bf16* __restrict__ alo) {
    const int wave = threadIdx.x >> 5;
    const int lane = threadIdx.x & 31;
    const int row  = blockIdx.x * 8 + wave;
    const float* r = emb + (size_t)row * EMB;

    float x[16];
    float s = 0.f;
#pragma unroll
    for (int i = 0; i < 16; ++i) { x[i] = r[lane + 32 * i]; s += x[i] * x[i]; }
    s = wave_reduce_sum(s);
    const float inv = 1.0f / fmaxf(sqrtf(s), 1e-12f);

#pragma unroll
    for (int i = 0; i < 16; ++i) {
        const float xn = x[i] * inv;
        const __bf16 h = (__bf16)xn;
        const __bf16 l = (__bf16)(xn - (float)h);
        ahi[(size_t)row * EMB + lane + 32 * i] = h;
        alo[(size_t)row * EMB + lane + 32 * i] = l;
    }
}

// ---------------------------------------------------------------------------
// Kernel 2: fused weight-normalize + split-bf16 WMMA GEMM + ArcFace epilogue.
// Block tile: 32 classes x 512 batch rows (weight streamed from HBM once).
// 8 waves, each computing a 64(M) x 32(N) tile = 4x2 WMMA accumulators.
// cosine = Ahi*Bhi + Ahi*Blo + Alo*Bhi  (3x v_wmma_f32_16x16x32_bf16).
// ---------------------------------------------------------------------------
__global__ __launch_bounds__(256)
void arcface_gemm(const __bf16* __restrict__ ahi, const __bf16* __restrict__ alo,
                  const float* __restrict__ weight, const int* __restrict__ labels,
                  float* __restrict__ out) {
    __shared__ __align__(16) __bf16 Bh[32][LDK];
    __shared__ __align__(16) __bf16 Bl[32][LDK];

    const int tid   = threadIdx.x;
    const int wave  = tid >> 5;
    const int lane  = tid & 31;
    const int hf    = lane >> 4;   // lane half (0/1)
    const int hr    = lane & 15;   // row/col within half
    const int nBase = blockIdx.x * 32;

    // --- per-wave: compute 1/||w|| for its 4 owned weight rows -------------
    float inv[4];
#pragma unroll
    for (int j = 0; j < 4; ++j) {
        const int n = wave * 4 + j;
        const float* wr = weight + (size_t)(nBase + n) * EMB;
        float s = 0.f;
#pragma unroll
        for (int i = 0; i < 16; ++i) { const float v = wr[lane + 32 * i]; s += v * v; }
        s = wave_reduce_sum(s);
        inv[j] = 1.0f / fmaxf(sqrtf(s), 1e-12f);
    }

    const v8f vzero = {0.f, 0.f, 0.f, 0.f, 0.f, 0.f, 0.f, 0.f};
    v8f acc[4][2];
#pragma unroll
    for (int mi = 0; mi < 4; ++mi)
#pragma unroll
        for (int ni = 0; ni < 2; ++ni) acc[mi][ni] = vzero;

    // --- K loop in two 256-wide phases staged through LDS ------------------
    for (int p = 0; p < 2; ++p) {
        // stage: normalize + hi/lo split 32 weight rows into LDS
#pragma unroll
        for (int j = 0; j < 4; ++j) {
            const int n = wave * 4 + j;
            const float* wr = weight + (size_t)(nBase + n) * EMB + p * KC + lane * 8;
            const float4 f0 = *(const float4*)(wr);
            const float4 f1 = *(const float4*)(wr + 4);
            const float xs[8] = {f0.x, f0.y, f0.z, f0.w, f1.x, f1.y, f1.z, f1.w};
            v8bf h, l;
#pragma unroll
            for (int e = 0; e < 8; ++e) {
                const float xn = xs[e] * inv[j];
                const __bf16 hh = (__bf16)xn;
                h[e] = hh;
                l[e] = (__bf16)(xn - (float)hh);
            }
            *(v8bf*)&Bh[n][lane * 8] = h;
            *(v8bf*)&Bl[n][lane * 8] = l;
        }
        __syncthreads();

        for (int kb = 0; kb < KC; kb += 32) {
            // B fragments (contiguous K layout: lane<16 -> K 0..15, lane>=16 -> K 16..31)
            v16bf bh[2], bl[2];
#pragma unroll
            for (int ni = 0; ni < 2; ++ni) {
                const __bf16* bp = &Bh[ni * 16 + hr][kb + hf * 16];
                const v8bf b0 = *(const v8bf*)bp;
                const v8bf b1 = *(const v8bf*)(bp + 8);
                bh[ni] = __builtin_shufflevector(b0, b1, 0, 1, 2, 3, 4, 5, 6, 7,
                                                 8, 9, 10, 11, 12, 13, 14, 15);
                const __bf16* lp = &Bl[ni * 16 + hr][kb + hf * 16];
                const v8bf l0 = *(const v8bf*)lp;
                const v8bf l1 = *(const v8bf*)(lp + 8);
                bl[ni] = __builtin_shufflevector(l0, l1, 0, 1, 2, 3, 4, 5, 6, 7,
                                                 8, 9, 10, 11, 12, 13, 14, 15);
            }
#pragma unroll
            for (int mi = 0; mi < 4; ++mi) {
                // A fragments (contiguous K layout: lane<16 -> K 0..15, lane>=16 -> K 16..31)
                const int row = wave * 64 + mi * 16 + hr;
                const size_t base = (size_t)row * EMB + p * KC + kb + hf * 16;
                const v8bf a0 = *(const v8bf*)(ahi + base);
                const v8bf a1 = *(const v8bf*)(ahi + base + 8);
                const v16bf ah = __builtin_shufflevector(a0, a1, 0, 1, 2, 3, 4, 5, 6, 7,
                                                         8, 9, 10, 11, 12, 13, 14, 15);
                const v8bf c0 = *(const v8bf*)(alo + base);
                const v8bf c1 = *(const v8bf*)(alo + base + 8);
                const v16bf al = __builtin_shufflevector(c0, c1, 0, 1, 2, 3, 4, 5, 6, 7,
                                                         8, 9, 10, 11, 12, 13, 14, 15);
#pragma unroll
                for (int ni = 0; ni < 2; ++ni) {
                    acc[mi][ni] = __builtin_amdgcn_wmma_f32_16x16x32_bf16(
                        false, ah, false, bh[ni], (short)0, acc[mi][ni], false, false);
                    acc[mi][ni] = __builtin_amdgcn_wmma_f32_16x16x32_bf16(
                        false, ah, false, bl[ni], (short)0, acc[mi][ni], false, false);
                    acc[mi][ni] = __builtin_amdgcn_wmma_f32_16x16x32_bf16(
                        false, al, false, bh[ni], (short)0, acc[mi][ni], false, false);
                }
            }
        }
        __syncthreads();
    }

    // --- ArcFace epilogue: phi/one-hot transform + scale, single store -----
#pragma unroll
    for (int mi = 0; mi < 4; ++mi) {
        const int rbase = wave * 64 + mi * 16 + hf * 8;
#pragma unroll
        for (int ni = 0; ni < 2; ++ni) {
            const int col = nBase + ni * 16 + hr;
#pragma unroll
            for (int i = 0; i < 8; ++i) {
                const int row = rbase + i;
                const float c  = acc[mi][ni][i];
                const float sn = sqrtf(fmaxf(1.0f - c * c, 0.0f));
                const float phi = c * ARC_CM - sn * ARC_SM;
                const float pv  = (c > ARC_TH) ? phi : (c - ARC_MM);
                const int lbl = labels[row];
                const float v = ((lbl == col) ? pv : c) * ARC_S;
                out[1 + (size_t)row * NCLS + col] = v;
            }
        }
    }
}

// ---------------------------------------------------------------------------
// Kernel 3: per-row online softmax (single sweep) -> rowloss[b]
// ---------------------------------------------------------------------------
__global__ __launch_bounds__(256)
void row_softmax_loss(const float* __restrict__ out, const int* __restrict__ labels,
                      float* __restrict__ rowloss) {
    const int b = blockIdx.x;
    const float* row = out + 1 + (size_t)b * NCLS;

    float m = -INFINITY, s = 0.f;
    for (int i = threadIdx.x; i < NCLS; i += 256) {
        const float x  = row[i];
        const float nm = fmaxf(m, x);
        s = s * expf(m - nm) + expf(x - nm);
        m = nm;
    }

    __shared__ float sm[256];
    __shared__ float ss[256];
    sm[threadIdx.x] = m;
    ss[threadIdx.x] = s;
    __syncthreads();
    for (int off = 128; off > 0; off >>= 1) {
        if (threadIdx.x < off) {
            const float m1 = sm[threadIdx.x], s1 = ss[threadIdx.x];
            const float m2 = sm[threadIdx.x + off], s2 = ss[threadIdx.x + off];
            const float nm = fmaxf(m1, m2);
            sm[threadIdx.x] = nm;
            ss[threadIdx.x] = s1 * expf(m1 - nm) + s2 * expf(m2 - nm);
        }
        __syncthreads();
    }
    if (threadIdx.x == 0) {
        const float lse = sm[0] + logf(ss[0]);
        const int lbl = labels[b];
        rowloss[b] = lse - row[lbl];
    }
}

// ---------------------------------------------------------------------------
// Kernel 4: deterministic mean of 512 row losses -> d_out[0]
// ---------------------------------------------------------------------------
__global__ __launch_bounds__(256)
void finalize_loss(const float* __restrict__ rowloss, float* __restrict__ out) {
    __shared__ float s[256];
    s[threadIdx.x] = rowloss[threadIdx.x] + rowloss[threadIdx.x + 256];
    __syncthreads();
    for (int off = 128; off > 0; off >>= 1) {
        if (threadIdx.x < off) s[threadIdx.x] += s[threadIdx.x + off];
        __syncthreads();
    }
    if (threadIdx.x == 0) out[0] = s[0] / (float)BATCH;
}

// ---------------------------------------------------------------------------
extern "C" void kernel_launch(void* const* d_in, const int* in_sizes, int n_in,
                              void* d_out, int out_size, void* d_ws, size_t ws_size,
                              hipStream_t stream) {
    (void)in_sizes; (void)n_in; (void)out_size; (void)ws_size;
    const float* emb    = (const float*)d_in[0];
    const int*   labels = (const int*)d_in[1];
    const float* weight = (const float*)d_in[2];
    float* out = (float*)d_out;

    __bf16* ahi = (__bf16*)d_ws;                        // 512*512 bf16
    __bf16* alo = ahi + (size_t)BATCH * EMB;            // 512*512 bf16
    float*  rowloss = (float*)(alo + (size_t)BATCH * EMB);  // 512 f32

    norm_split_emb<<<BATCH / 8, 256, 0, stream>>>(emb, ahi, alo);
    arcface_gemm<<<NCLS / 32, 256, 0, stream>>>(ahi, alo, weight, labels, out);
    row_softmax_loss<<<BATCH, 256, 0, stream>>>(out, labels, rowloss);
    finalize_loss<<<1, 256, 0, stream>>>(rowloss, out);
}
